// MultiHeadAttention_6451040878907
// MI455X (gfx1250) — matrix operand order repaired and checked by reading, not compiled, not run on hardware
//
#include <hip/hip_runtime.h>
#include <hip/hip_bf16.h>

typedef __bf16 v16bf __attribute__((ext_vector_type(16)));
typedef float  v8f   __attribute__((ext_vector_type(8)));
typedef unsigned int v4u __attribute__((ext_vector_type(4)));
typedef int v4i __attribute__((ext_vector_type(4)));

#define SEQ    1024
#define DMODEL 1024
#define NHEAD  16
#define HD     64
#define BATCH  4

#if __has_builtin(__builtin_amdgcn_global_load_async_to_lds_b128) && \
    __has_builtin(__builtin_amdgcn_s_wait_asynccnt)
#define USE_ASYNC_LDS 1
#endif

// A-matrix (16x32 bf16) element->K map per ISA 7.12.2:
// lanes 0-15: VGPR0-3 hold K=0..7, VGPR4-7 hold K=16..23
// lanes16-31: VGPR0-3 hold K=8..15, VGPR4-7 hold K=24..31
__device__ __forceinline__ int a_kmap(int i, int half) {
  return (i < 8) ? (half * 8 + i) : (16 + half * 8 + (i - 8));
}

__global__ void f32_to_bf16_kernel(const float* __restrict__ in,
                                   __bf16* __restrict__ out, int n) {
  int i = blockIdx.x * blockDim.x + threadIdx.x;
  if (i < n) out[i] = (__bf16)in[i];
}

// in: [R][C] f32 row-major -> out: [C][R] bf16 (transposed weights)
__global__ void f32_to_bf16_T_kernel(const float* __restrict__ in,
                                     __bf16* __restrict__ out, int R, int C) {
  int i = blockIdx.x * blockDim.x + threadIdx.x;
  if (i < R * C) {
    const int r = i / C;
    const int c = i - r * C;
    out[(size_t)c * R + r] = (__bf16)in[i];
  }
}

// Copy one 16-byte chunk global -> LDS (async engine path when available).
__device__ __forceinline__ void stage16(const __bf16* gsrc, __bf16* ldst) {
#ifdef USE_ASYNC_LDS
  __builtin_amdgcn_global_load_async_to_lds_b128(
      (__attribute__((address_space(1))) v4i*)(unsigned long long)(const void*)gsrc,
      (__attribute__((address_space(3))) v4i*)(unsigned long long)(const void*)ldst,
      0, 0);
#else
  *(v4u*)ldst = *(const v4u*)gsrc;
#endif
}

__device__ __forceinline__ void stage_wait() {
#ifdef USE_ASYNC_LDS
  __builtin_amdgcn_s_wait_asynccnt(0);
#endif
}

// C[M,N] = A[M,K](bf16) * Bt[N,K]^T(bf16) + bias[N].
// 256 threads = 8 waves: 4 row-slabs x 2 col-slabs; macro tile 64 x 128.
// Each wave: 16 rows x 64 cols = 4 WMMA accumulators, A frag shared.
// A tile (64x32) and Bt tile (128x32) double-buffered in LDS via the async
// global->LDS engine; fragments gathered from LDS as contiguous b128 pairs.
template <bool OUT_BF16>
__global__ void gemm_bf16_wmma(const __bf16* __restrict__ A,
                               const __bf16* __restrict__ Bt,
                               const float* __restrict__ bias,
                               void* __restrict__ C,
                               int M, int N, int K) {
  __shared__ __bf16 Abuf[2][64 * 32];   // 2 x 4KB
  __shared__ __bf16 Bbuf[2][128 * 32];  // 2 x 8KB

  const int tid   = threadIdx.x;
  const int lane  = tid & 31;
  const int wave  = tid >> 5;
  const int half  = lane >> 4;
  const int nlow  = lane & 15;
  const int rslab = wave & 3;
  const int cslab = wave >> 2;

  const int row0  = blockIdx.y * 64;
  const int rbase = row0 + rslab * 16;
  const int colg0 = blockIdx.x * 128;          // macro-tile first column
  const int col0  = colg0 + cslab * 64;        // this wave's first column

  // Staging: A tile 64x32 = 256 chunks of 16B (1/thread);
  //          Bt tile 128x32 = 512 chunks of 16B (2/thread).
  const int astr = tid >> 2;         // A tile row 0..63
  const int astc = (tid & 3) * 8;    // A k-offset 0/8/16/24
  const int bc0  = tid;              // B chunks tid and tid+256
  const int bcol0 = bc0 >> 2,  bk0off = (bc0 & 3) * 8;
  const int bc1  = tid + 256;
  const int bcol1 = bc1 >> 2,  bk1off = (bc1 & 3) * 8;

  const int nIter = K / 32;
  stage16(&A[(size_t)(row0 + astr) * K + astc], &Abuf[0][astr * 32 + astc]);
  stage16(&Bt[(size_t)(colg0 + bcol0) * K + bk0off], &Bbuf[0][bcol0 * 32 + bk0off]);
  stage16(&Bt[(size_t)(colg0 + bcol1) * K + bk1off], &Bbuf[0][bcol1 * 32 + bk1off]);

  v8f acc0 = {}, acc1 = {}, acc2 = {}, acc3 = {};
  for (int it = 0; it < nIter; ++it) {
    const int cur = it & 1;
    stage_wait();
    __syncthreads();
    if (it + 1 < nIter) {
      const int kn = (it + 1) * 32;
      stage16(&A[(size_t)(row0 + astr) * K + kn + astc],
              &Abuf[1 - cur][astr * 32 + astc]);
      stage16(&Bt[(size_t)(colg0 + bcol0) * K + kn + bk0off],
              &Bbuf[1 - cur][bcol0 * 32 + bk0off]);
      stage16(&Bt[(size_t)(colg0 + bcol1) * K + kn + bk1off],
              &Bbuf[1 - cur][bcol1 * 32 + bk1off]);
    }

    // A fragment: two contiguous 16B chunks per lane from LDS.
    const __bf16* lrowA = &Abuf[cur][(rslab * 16 + nlow) * 32];
    v16bf a;
#pragma unroll
    for (int i = 0; i < 8; ++i) a[i] = lrowA[half * 8 + i];
#pragma unroll
    for (int i = 0; i < 8; ++i) a[8 + i] = lrowA[16 + half * 8 + i];

    // B fragments: per-lane column read as two 16B chunks of the Bt tile (a_kmap k order).
    v16bf b0, b1, b2, b3;
    const int bl = cslab * 64 + nlow;  // lane's column within macro tile
#pragma unroll
    for (int i = 0; i < 16; ++i) {
      const int k = a_kmap(i, half);
      b0[i] = Bbuf[cur][(bl)      * 32 + k];
      b1[i] = Bbuf[cur][(bl + 16) * 32 + k];
      b2[i] = Bbuf[cur][(bl + 32) * 32 + k];
      b3[i] = Bbuf[cur][(bl + 48) * 32 + k];
    }
    acc0 = __builtin_amdgcn_wmma_f32_16x16x32_bf16(false, a, false, b0, (short)0, acc0, false, false);
    acc1 = __builtin_amdgcn_wmma_f32_16x16x32_bf16(false, a, false, b1, (short)0, acc1, false, false);
    acc2 = __builtin_amdgcn_wmma_f32_16x16x32_bf16(false, a, false, b2, (short)0, acc2, false, false);
    acc3 = __builtin_amdgcn_wmma_f32_16x16x32_bf16(false, a, false, b3, (short)0, acc3, false, false);
    __syncthreads();
  }

#pragma unroll
  for (int j = 0; j < 4; ++j) {
    const v8f acc = (j == 0) ? acc0 : (j == 1) ? acc1 : (j == 2) ? acc2 : acc3;
    const int bcol = col0 + j * 16 + nlow;
    const float bv = bias[bcol];
#pragma unroll
    for (int r = 0; r < 8; ++r) {
      const int orow = rbase + r + 8 * half;
      const float v = acc[r] + bv;
      if (OUT_BF16)
        ((__bf16*)C)[(size_t)orow * N + bcol] = (__bf16)v;
      else
        ((float*)C)[(size_t)orow * N + bcol] = v;
    }
  }
}

// One block per (query-row-block of 16, b*h). qkv layout: [B,S,3*D] with head h
// at column h*192 (+0 q, +64 k, +128 v). vals out: [B,S,D] bf16.
__global__ void attn_wmma_kernel(const __bf16* __restrict__ qkv,
                                 const int* __restrict__ mask,
                                 __bf16* __restrict__ vals) {
  extern __shared__ float smem[];
  float* sc = smem;                  // 16 x SEQ f32 score strip
  float* rowsum = smem + 16 * SEQ;   // 16 f32

  const int rb = blockIdx.x;
  const int b  = blockIdx.y >> 4;
  const int h  = blockIdx.y & 15;
  const int tid  = threadIdx.x;
  const int lane = tid & 31;
  const int wave = tid >> 5;
  const int half = lane >> 4;
  const int nlow = lane & 15;

  const size_t hbase = ((size_t)b * SEQ) * (3 * DMODEL) + (size_t)h * (3 * HD);
  const int qrow = rb * 16 + nlow;

  v16bf aq0, aq1;
#pragma unroll
  for (int i = 0; i < 16; ++i) {
    const int k0 = a_kmap(i, half);
    aq0[i] = qkv[hbase + (size_t)qrow * (3 * DMODEL) + k0];
    aq1[i] = qkv[hbase + (size_t)qrow * (3 * DMODEL) + 32 + k0];
  }

  // ---- Phase 1: scores = mask(QK^T / 8) into LDS ----
  for (int ct = wave; ct < SEQ / 16; ct += 4) {
    const int kcol = ct * 16 + nlow;
    v16bf bk0, bk1;
#pragma unroll
    for (int i = 0; i < 16; ++i) {
      const int kd = a_kmap(i, half);
      bk0[i] = qkv[hbase + (size_t)kcol * (3 * DMODEL) + HD + kd];
      bk1[i] = qkv[hbase + (size_t)kcol * (3 * DMODEL) + HD + 32 + kd];
    }
    v8f acc = {};
    acc = __builtin_amdgcn_wmma_f32_16x16x32_bf16(false, aq0, false, bk0, (short)0, acc, false, false);
    acc = __builtin_amdgcn_wmma_f32_16x16x32_bf16(false, aq1, false, bk1, (short)0, acc, false, false);
#pragma unroll
    for (int r = 0; r < 8; ++r) {
      const int m   = r + 8 * half;
      const int qr  = rb * 16 + m;
      const int col = ct * 16 + nlow;
      float s = acc[r] * 0.125f;  // 1/sqrt(64)
      if (mask[((size_t)b * SEQ + qr) * SEQ + col] == 0) s = -1e20f;
      sc[m * SEQ + col] = s;
    }
  }
  __syncthreads();

  // ---- Phase 2: softmax, 8 threads per row, butterfly reduce in-wave ----
  {
    const int row = tid >> 3;
    const int sub = tid & 7;
    float mx = -3.0e38f;
    for (int j = 0; j < SEQ / 8; ++j)
      mx = fmaxf(mx, sc[row * SEQ + sub + j * 8]);
    mx = fmaxf(mx, __shfl_xor(mx, 1, 8));
    mx = fmaxf(mx, __shfl_xor(mx, 2, 8));
    mx = fmaxf(mx, __shfl_xor(mx, 4, 8));
    float sum = 0.0f;
    for (int j = 0; j < SEQ / 8; ++j) {
      const int c = row * SEQ + sub + j * 8;
      const float e = __expf(sc[c] - mx);
      sc[c] = e;
      sum += e;
    }
    sum += __shfl_xor(sum, 1, 8);
    sum += __shfl_xor(sum, 2, 8);
    sum += __shfl_xor(sum, 4, 8);
    if (sub == 0) rowsum[row] = sum;
  }
  __syncthreads();

  // ---- Phase 3: P @ V; each wave owns 16 output columns of hd=64 ----
  const float rinv = 1.0f / rowsum[nlow];
  const int dcol = wave * 16 + nlow;
  v8f acc = {};
  for (int kt = 0; kt < SEQ; kt += 32) {
    v16bf a, bv;
#pragma unroll
    for (int i = 0; i < 16; ++i) {
      a[i] = (__bf16)(sc[nlow * SEQ + kt + a_kmap(i, half)] * rinv);
      const int kv = kt + a_kmap(i, half);
      bv[i] = qkv[hbase + (size_t)kv * (3 * DMODEL) + 2 * HD + dcol];
    }
    acc = __builtin_amdgcn_wmma_f32_16x16x32_bf16(false, a, false, bv, (short)0, acc, false, false);
  }
#pragma unroll
  for (int r = 0; r < 8; ++r) {
    const int srow = rb * 16 + r + 8 * half;
    vals[((size_t)b * SEQ + srow) * DMODEL + h * HD + dcol] = (__bf16)acc[r];
  }
}

extern "C" void kernel_launch(void* const* d_in, const int* in_sizes, int n_in,
                              void* d_out, int out_size, void* d_ws, size_t ws_size,
                              hipStream_t stream) {
  const float* x    = (const float*)d_in[0];
  const int*   mask = (const int*)d_in[1];
  const float* Wqkv = (const float*)d_in[2];
  const float* bqkv = (const float*)d_in[3];
  const float* Wo   = (const float*)d_in[4];
  const float* bo   = (const float*)d_in[5];
  float* out = (float*)d_out;

  const int n_x    = BATCH * SEQ * DMODEL;
  const int n_wqkv = DMODEL * 3 * DMODEL;
  const int n_wo   = DMODEL * DMODEL;
  const size_t n_qkv = (size_t)BATCH * SEQ * 3 * DMODEL;
  __bf16* x_bf     = (__bf16*)d_ws;
  __bf16* wqkvT_bf = x_bf + n_x;        // [3D][D] transposed
  __bf16* woT_bf   = wqkvT_bf + n_wqkv; // [D][D] transposed
  __bf16* qkv_bf   = woT_bf + n_wo;
  __bf16* vals_bf  = qkv_bf + n_qkv;

  f32_to_bf16_kernel<<<(n_x + 255) / 256, 256, 0, stream>>>(x, x_bf, n_x);
  f32_to_bf16_T_kernel<<<(n_wqkv + 255) / 256, 256, 0, stream>>>(
      Wqkv, wqkvT_bf, DMODEL, 3 * DMODEL);
  f32_to_bf16_T_kernel<<<(n_wo + 255) / 256, 256, 0, stream>>>(
      Wo, woT_bf, DMODEL, DMODEL);

  // QKV projection: [4096,1024] x [1024,3072] -> bf16 qkv
  gemm_bf16_wmma<true><<<dim3(3 * DMODEL / 128, BATCH * SEQ / 64), 256, 0, stream>>>(
      x_bf, wqkvT_bf, bqkv, qkv_bf, BATCH * SEQ, 3 * DMODEL, DMODEL);

  // Attention: 64 row-blocks x (B*H)=64
  const size_t smem_bytes = (16 * SEQ + 16) * sizeof(float);
  attn_wmma_kernel<<<dim3(SEQ / 16, BATCH * NHEAD), 128, smem_bytes, stream>>>(
      qkv_bf, mask, vals_bf);

  // Output projection: [4096,1024] x [1024,1024] -> f32 out
  gemm_bf16_wmma<false><<<dim3(DMODEL / 128, BATCH * SEQ / 64), 256, 0, stream>>>(
      vals_bf, woT_bf, bo, out, BATCH * SEQ, DMODEL, DMODEL);
}
